// SatelliteDensityModelV4_78323023610134
// MI455X (gfx1250) — matrix operand order repaired and checked by reading, not compiled, not run on hardware
//
#include <hip/hip_runtime.h>
#include <hip/hip_bf16.h>
#include <math.h>

// ---------------------------------------------------------------------------
// SatelliteDensityModelV4 for MI455X (gfx1250, wave32, WMMA)
//
// Pipeline:
//   prep:  f32->f16 transposed weights, f32->f16 omni_seq
//   static: static MLP + LN + gates + gated phys
//   attn:  qkv -> online-softmax per-channel attention -> out proj
//   conv:  3 dilated conv1d + gelu + mean
//   scan0: bidirectional GRU layer 0 (WMMA per step), writes x1 (f16)
//   scan1: bidirectional GRU layer 1 (WMMA per step), writes omni_embed
//   head:  concat + LN + MLP -> (B,432)
// ---------------------------------------------------------------------------

typedef _Float16 v16h __attribute__((ext_vector_type(16)));
typedef _Float16 h8   __attribute__((ext_vector_type(8)));
typedef float    v8f  __attribute__((ext_vector_type(8)));

#define B_    128
#define T_    1024
#define OMNI  32
#define H_    64
#define G3    192      // 3*H
#define NP    5
#define PL    432
#define HIDD  128
#define FUSED 288      // 128 + 64 + 96

__device__ __forceinline__ float gelu_exact(float x) {
    return 0.5f * x * (1.0f + erff(x * 0.70710678118654752f));
}
__device__ __forceinline__ float sigmoidf_(float x) {
    return 1.0f / (1.0f + __expf(-x));
}

// ------------------------- prep kernels ------------------------------------

// W (N x K) f32  ->  WT (K x N) f16
__global__ void k_transpose_f16(const float* __restrict__ W,
                                _Float16* __restrict__ WT, int N, int K) {
    int idx = blockIdx.x * 256 + threadIdx.x;
    if (idx < N * K) {
        int n = idx / K, k = idx % K;
        WT[(size_t)k * N + n] = (_Float16)W[(size_t)n * K + k];
    }
}

__global__ void k_f32_to_f16(const float* __restrict__ src,
                             _Float16* __restrict__ dst, int n) {
    int idx = blockIdx.x * 256 + threadIdx.x;
    if (idx < n) dst[idx] = (_Float16)src[idx];
}

// ------------------------- GRU scan (WMMA) ---------------------------------
// One block = 16 batch rows, one (layer,direction). 128 threads = 4 waves.
// Per step: gi = x_t @ WihT (K=DIN), gh = h @ WhhT (K=64), both via
// v_wmma_f32_16x16x32_f16, staged to LDS, then fused GRU gate update.
template <int DIN, bool WRITE_SEQ, bool WRITE_HT>
__global__ __launch_bounds__(128)
void k_gru_scan(const _Float16* __restrict__ xin,        // [B][T][DIN]
                const _Float16* __restrict__ WihT_f,     // [DIN][192]
                const _Float16* __restrict__ WihT_b,
                const _Float16* __restrict__ WhhT_f,     // [64][192]
                const _Float16* __restrict__ WhhT_b,
                const float* __restrict__ bih_f, const float* __restrict__ bih_b,
                const float* __restrict__ bhh_f, const float* __restrict__ bhh_b,
                _Float16* __restrict__ yout,             // [B][T][128] (x1)
                float* __restrict__ ht_out)              // [B][128]
{
    const int dir = blockIdx.y;              // 0 = forward, 1 = reversed
    const int b0  = blockIdx.x * 16;
    const _Float16* WihT = dir ? WihT_b : WihT_f;
    const _Float16* WhhT = dir ? WhhT_b : WhhT_f;
    const float*    bih  = dir ? bih_b  : bih_f;
    const float*    bhh  = dir ? bhh_b  : bhh_f;

    __shared__ float     s_gi[16][G3];
    __shared__ float     s_gh[16][G3];
    __shared__ _Float16  s_h[16][H_];
    __shared__ float     s_bih[G3];
    __shared__ float     s_bhh[G3];

    const int tid  = threadIdx.x;
    const int lane = tid & 31;
    const int wave = tid >> 5;
    const int m    = lane & 15;              // A-matrix row / C-matrix col
    const int kh   = (lane >> 4) << 3;       // 0 or 8 : A layout K offset
    const int mhi  = (lane >> 4) << 3;       // 0 or 8 : C layout M offset

    for (int i = tid; i < G3; i += 128) { s_bih[i] = bih[i]; s_bhh[i] = bhh[i]; }
    for (int i = tid; i < 16 * H_; i += 128) (&s_h[0][0])[i] = (_Float16)0.0f;

    float hreg[8];
    #pragma unroll
    for (int u = 0; u < 8; ++u) hreg[u] = 0.0f;
    __syncthreads();

    for (int tt = 0; tt < T_; ++tt) {
        const int t = dir ? (T_ - 1 - tt) : tt;
        const _Float16* xrow = xin + ((size_t)(b0 + m) * T_ + t) * DIN;

        // software prefetch next timestep's activation row (uniform guard)
        if (tt + 1 < T_) {
            const int tn = dir ? (t - 1) : (t + 1);
            __builtin_prefetch(xin + ((size_t)(b0 + m) * T_ + tn) * DIN, 0, 0);
        }

        #pragma unroll
        for (int i = 0; i < 3; ++i) {
            const int nb = (wave * 3 + i) * 16;

            // ---- gi tile: x_t (16 x DIN) @ WihT -> (16 x 16) ----
            v8f acc = {};
            #pragma unroll
            for (int kb = 0; kb < DIN; kb += 32) {
                union { v16h v; h8 h[2]; } A;
                A.h[0] = *(const h8*)(xrow + kb + kh);
                A.h[1] = *(const h8*)(xrow + kb + kh + 16);
                v16h Bm;
                #pragma unroll
                for (int s = 0; s < 16; ++s)
                    Bm[s] = WihT[(size_t)(kb + kh + (s & 7) + ((s >> 3) << 4)) * G3 + nb + m];
                acc = __builtin_amdgcn_wmma_f32_16x16x32_f16(
                          false, A.v, false, Bm, (short)0, acc, false, false);
            }
            #pragma unroll
            for (int j = 0; j < 8; ++j) s_gi[j + mhi][nb + m] = acc[j];

            // ---- gh tile: h (16 x 64) @ WhhT -> (16 x 16) ----
            v8f acch = {};
            #pragma unroll
            for (int kb = 0; kb < H_; kb += 32) {
                union { v16h v; h8 h[2]; } A;
                A.h[0] = *(const h8*)(&s_h[m][kb + kh]);
                A.h[1] = *(const h8*)(&s_h[m][kb + kh + 16]);
                v16h Bm;
                #pragma unroll
                for (int s = 0; s < 16; ++s)
                    Bm[s] = WhhT[(size_t)(kb + kh + (s & 7) + ((s >> 3) << 4)) * G3 + nb + m];
                acch = __builtin_amdgcn_wmma_f32_16x16x32_f16(
                           false, A.v, false, Bm, (short)0, acch, false, false);
            }
            #pragma unroll
            for (int j = 0; j < 8; ++j) s_gh[j + mhi][nb + m] = acch[j];
        }
        __syncthreads();

        // ---- fused GRU gates: 16*64 updates over 128 threads ----
        #pragma unroll
        for (int u = 0; u < 8; ++u) {
            const int idx = tid * 8 + u;
            const int mm  = idx >> 6;
            const int j   = idx & 63;
            const float gr = s_gi[mm][j]      + s_bih[j]      + s_gh[mm][j]      + s_bhh[j];
            const float gz = s_gi[mm][64 + j] + s_bih[64 + j] + s_gh[mm][64 + j] + s_bhh[64 + j];
            const float r  = sigmoidf_(gr);
            const float z  = sigmoidf_(gz);
            const float n  = tanhf(s_gi[mm][128 + j] + s_bih[128 + j] +
                                   r * (s_gh[mm][128 + j] + s_bhh[128 + j]));
            const float hn = (1.0f - z) * n + z * hreg[u];
            hreg[u] = hn;
            s_h[mm][j] = (_Float16)hn;
            if (WRITE_SEQ)
                yout[((size_t)(b0 + mm) * T_ + t) * 128 + dir * H_ + j] = (_Float16)hn;
        }
        __syncthreads();
    }

    if (WRITE_HT) {
        #pragma unroll
        for (int u = 0; u < 8; ++u) {
            const int idx = tid * 8 + u;
            const int mm  = idx >> 6;
            const int j   = idx & 63;
            ht_out[(size_t)(b0 + mm) * 128 + dir * H_ + j] = hreg[u];
        }
    }
}

// ------------------------- static path -------------------------------------
// per batch row: static_embed = LN(gelu(x @ sm_W^T + b)); gates; p = phys*gates
__global__ __launch_bounds__(64)
void k_static(const float* __restrict__ static_x, const float* __restrict__ sm_W,
              const float* __restrict__ sm_b, const float* __restrict__ ln_g,
              const float* __restrict__ ln_b, const float* __restrict__ gate_W,
              const float* __restrict__ gate_b, const float* __restrict__ phys,
              float* __restrict__ static_embed, float* __restrict__ p)
{
    const int b = blockIdx.x;
    const int j = threadIdx.x;                 // 0..63
    __shared__ float s_x[16];
    __shared__ float s_g[64];
    __shared__ float s_e[64];
    __shared__ float s_gate[NP];

    if (j < 16) s_x[j] = static_x[b * 16 + j];
    __syncthreads();

    float acc = sm_b[j];
    for (int k = 0; k < 16; ++k) acc += s_x[k] * sm_W[j * 16 + k];
    const float g = gelu_exact(acc);
    s_g[j] = g;
    __syncthreads();

    float mean = 0.0f;
    for (int k = 0; k < 64; ++k) mean += s_g[k];
    mean *= (1.0f / 64.0f);
    float var = 0.0f;
    for (int k = 0; k < 64; ++k) { const float d = s_g[k] - mean; var += d * d; }
    var *= (1.0f / 64.0f);
    const float se = (g - mean) * rsqrtf(var + 1e-5f) * ln_g[j] + ln_b[j];
    static_embed[b * 64 + j] = se;
    s_e[j] = se;
    __syncthreads();

    if (j < NP) {
        float a = gate_b[j];
        for (int k = 0; k < 64; ++k) a += s_e[k] * gate_W[j * 64 + k];
        s_gate[j] = sigmoidf_(a);
    }
    __syncthreads();

    for (int idx = j; idx < NP * PL; idx += 64) {
        const int g2 = idx / PL, l = idx % PL;
        p[((size_t)b * NP + g2) * PL + l] =
            phys[((size_t)b * NP + g2) * PL + l] * s_gate[g2];
    }
}

// ------------------------- attention ---------------------------------------
__global__ void k_qkv(const float* __restrict__ p, const float* __restrict__ Wi,
                      const float* __restrict__ bi, float* __restrict__ qkv) {
    const int idx = blockIdx.x * 256 + threadIdx.x;
    if (idx >= B_ * PL) return;
    const int b = idx / PL, l = idx % PL;
    float pg[NP];
    #pragma unroll
    for (int g = 0; g < NP; ++g) pg[g] = p[((size_t)b * NP + g) * PL + l];
    #pragma unroll
    for (int o = 0; o < 3 * NP; ++o) {
        float a = bi[o];
        #pragma unroll
        for (int g = 0; g < NP; ++g) a += pg[g] * Wi[o * NP + g];
        qkv[((size_t)b * PL + l) * (3 * NP) + o] = a;
    }
}

// online-softmax rank-1 attention per (b, channel h, query i)
__global__ void k_attn(const float* __restrict__ qkv, float* __restrict__ ao) {
    const int idx = blockIdx.x * 256 + threadIdx.x;
    if (idx >= B_ * NP * PL) return;
    const int b = idx / (NP * PL);
    const int r = idx % (NP * PL);
    const int h = r / PL;
    const int i = r % PL;

    const float qv = qkv[((size_t)b * PL + i) * (3 * NP) + h];
    float mmax = -3.4e38f, lsum = 0.0f, acc = 0.0f;
    for (int j = 0; j < PL; ++j) {
        const float kv = qkv[((size_t)b * PL + j) * (3 * NP) + NP + h];
        const float vv = qkv[((size_t)b * PL + j) * (3 * NP) + 2 * NP + h];
        const float s = qv * kv;
        if (s > mmax) {
            const float c = __expf(mmax - s);
            lsum *= c; acc *= c; mmax = s;
        }
        const float e = __expf(s - mmax);
        lsum += e; acc += e * vv;
    }
    ao[((size_t)b * PL + i) * NP + h] = acc / lsum;
}

__global__ void k_attn_out(const float* __restrict__ ao, const float* __restrict__ Wo,
                           const float* __restrict__ bo, float* __restrict__ p2) {
    const int idx = blockIdx.x * 256 + threadIdx.x;
    if (idx >= B_ * PL) return;
    const int b = idx / PL, i = idx % PL;
    float a5[NP];
    #pragma unroll
    for (int h = 0; h < NP; ++h) a5[h] = ao[((size_t)b * PL + i) * NP + h];
    #pragma unroll
    for (int o = 0; o < NP; ++o) {
        float v = bo[o];
        #pragma unroll
        for (int h = 0; h < NP; ++h) v += a5[h] * Wo[o * NP + h];
        p2[((size_t)b * NP + o) * PL + i] = v;
    }
}

// ------------------------- dilated convs + mean ----------------------------
__global__ void k_conv(const float* __restrict__ p2,
                       const float* __restrict__ W1, const float* __restrict__ b1,
                       const float* __restrict__ W2, const float* __restrict__ b2,
                       const float* __restrict__ W3, const float* __restrict__ b3,
                       float* __restrict__ cfeat) {
    const int idx = blockIdx.x * 256 + threadIdx.x;
    if (idx >= B_ * 96) return;
    const int b = idx / 96, oc = idx % 96;
    const int which = oc / 32, ocl = oc % 32;
    const int ksz = (which == 0) ? 3 : (which == 1) ? 5 : 9;
    const int pad = (which == 0) ? 1 : (which == 1) ? 2 : 4;
    const int dil = (which == 0) ? 1 : (which == 1) ? 2 : 4;
    const float* W = (which == 0) ? W1 : (which == 1) ? W2 : W3;
    const float* bb = (which == 0) ? b1 : (which == 1) ? b2 : b3;
    const int Lout = PL + 2 * pad - dil * (ksz - 1);

    float sum = 0.0f;
    for (int l = 0; l < Lout; ++l) {
        float y = bb[ocl];
        for (int ic = 0; ic < NP; ++ic) {
            const float* row = p2 + ((size_t)b * NP + ic) * PL;
            for (int kk = 0; kk < ksz; ++kk) {
                const int pos = l + kk * dil - pad;
                if (pos >= 0 && pos < PL)
                    y += W[(ocl * NP + ic) * ksz + kk] * row[pos];
            }
        }
        sum += gelu_exact(y);
    }
    cfeat[(size_t)b * 96 + oc] = sum / (float)Lout;
}

// ------------------------- head --------------------------------------------
__global__ __launch_bounds__(128)
void k_head(const float* __restrict__ omni_embed, const float* __restrict__ static_embed,
            const float* __restrict__ cfeat, const float* __restrict__ ln_g,
            const float* __restrict__ ln_b, const float* __restrict__ W1,
            const float* __restrict__ b1, const float* __restrict__ W2,
            const float* __restrict__ b2, float* __restrict__ out) {
    const int b = blockIdx.x;
    const int tid = threadIdx.x;
    __shared__ float s_f[FUSED];
    __shared__ float s_h1[HIDD];

    for (int i = tid; i < FUSED; i += 128) {
        float v;
        if (i < 128)      v = omni_embed[(size_t)b * 128 + i];
        else if (i < 192) v = static_embed[(size_t)b * 64 + (i - 128)];
        else              v = cfeat[(size_t)b * 96 + (i - 192)];
        s_f[i] = v;
    }
    __syncthreads();

    float mean = 0.0f;
    for (int k = 0; k < FUSED; ++k) mean += s_f[k];
    mean *= (1.0f / (float)FUSED);
    float var = 0.0f;
    for (int k = 0; k < FUSED; ++k) { const float d = s_f[k] - mean; var += d * d; }
    var *= (1.0f / (float)FUSED);
    const float inv = rsqrtf(var + 1e-5f);
    __syncthreads();
    for (int i = tid; i < FUSED; i += 128)
        s_f[i] = (s_f[i] - mean) * inv * ln_g[i] + ln_b[i];
    __syncthreads();

    {
        float acc = b1[tid];
        for (int k = 0; k < FUSED; ++k) acc += s_f[k] * W1[tid * FUSED + k];
        s_h1[tid] = gelu_exact(acc);
    }
    __syncthreads();

    for (int r = tid; r < PL; r += 128) {
        float acc = b2[r];
        for (int k = 0; k < HIDD; ++k) acc += s_h1[k] * W2[r * HIDD + k];
        out[(size_t)b * PL + r] = acc;
    }
}

// ------------------------- launcher ----------------------------------------
extern "C" void kernel_launch(void* const* d_in, const int* in_sizes, int n_in,
                              void* d_out, int out_size, void* d_ws, size_t ws_size,
                              hipStream_t stream) {
    const float* omni_seq = (const float*)d_in[0];
    const float* static_x = (const float*)d_in[1];
    const float* phys     = (const float*)d_in[2];
    const float* Wih[4]   = { (const float*)d_in[3],  (const float*)d_in[7],
                              (const float*)d_in[11], (const float*)d_in[15] };
    const float* Whh[4]   = { (const float*)d_in[4],  (const float*)d_in[8],
                              (const float*)d_in[12], (const float*)d_in[16] };
    const float* bih[4]   = { (const float*)d_in[5],  (const float*)d_in[9],
                              (const float*)d_in[13], (const float*)d_in[17] };
    const float* bhh[4]   = { (const float*)d_in[6],  (const float*)d_in[10],
                              (const float*)d_in[14], (const float*)d_in[18] };
    const float* sm_W = (const float*)d_in[19];
    const float* sm_b = (const float*)d_in[20];
    const float* sm_ln_g = (const float*)d_in[21];
    const float* sm_ln_b = (const float*)d_in[22];
    const float* gate_W = (const float*)d_in[23];
    const float* gate_b = (const float*)d_in[24];
    const float* attn_in_W  = (const float*)d_in[25];
    const float* attn_in_b  = (const float*)d_in[26];
    const float* attn_out_W = (const float*)d_in[27];
    const float* attn_out_b = (const float*)d_in[28];
    const float* conv1_W = (const float*)d_in[29];
    const float* conv1_b = (const float*)d_in[30];
    const float* conv2_W = (const float*)d_in[31];
    const float* conv2_b = (const float*)d_in[32];
    const float* conv3_W = (const float*)d_in[33];
    const float* conv3_b = (const float*)d_in[34];
    const float* head_ln_g = (const float*)d_in[35];
    const float* head_ln_b = (const float*)d_in[36];
    const float* head_W1 = (const float*)d_in[37];
    const float* head_b1 = (const float*)d_in[38];
    const float* head_W2 = (const float*)d_in[39];
    const float* head_b2 = (const float*)d_in[40];

    // ---- workspace layout ----
    char* base = (char*)d_ws;
    size_t off = 0;
    auto alloc = [&](size_t bytes) -> void* {
        void* p = base + off;
        off = (off + bytes + 255) & ~(size_t)255;
        return p;
    };
    _Float16* omni_h   = (_Float16*)alloc((size_t)B_ * T_ * OMNI * 2);
    _Float16* x1       = (_Float16*)alloc((size_t)B_ * T_ * 128 * 2);
    _Float16* WihT0[2] = { (_Float16*)alloc(OMNI * G3 * 2), (_Float16*)alloc(OMNI * G3 * 2) };
    _Float16* WhhT0[2] = { (_Float16*)alloc(H_ * G3 * 2),   (_Float16*)alloc(H_ * G3 * 2) };
    _Float16* WihT1[2] = { (_Float16*)alloc(128 * G3 * 2),  (_Float16*)alloc(128 * G3 * 2) };
    _Float16* WhhT1[2] = { (_Float16*)alloc(H_ * G3 * 2),   (_Float16*)alloc(H_ * G3 * 2) };
    float* omni_embed   = (float*)alloc((size_t)B_ * 128 * 4);
    float* static_embed = (float*)alloc((size_t)B_ * 64 * 4);
    float* p_buf  = (float*)alloc((size_t)B_ * NP * PL * 4);
    float* qkv    = (float*)alloc((size_t)B_ * PL * 3 * NP * 4);
    float* ao     = (float*)alloc((size_t)B_ * PL * NP * 4);
    float* p2     = (float*)alloc((size_t)B_ * NP * PL * 4);
    float* cfeat  = (float*)alloc((size_t)B_ * 96 * 4);
    (void)ws_size; (void)n_in; (void)in_sizes; (void)out_size;

    // ---- prep: transposed f16 weights ----
    struct { const float* W; _Float16* WT; int N; int K; } tr[8] = {
        { Wih[0], WihT0[0], G3, OMNI }, { Wih[1], WihT0[1], G3, OMNI },
        { Whh[0], WhhT0[0], G3, H_   }, { Whh[1], WhhT0[1], G3, H_   },
        { Wih[2], WihT1[0], G3, 128  }, { Wih[3], WihT1[1], G3, 128  },
        { Whh[2], WhhT1[0], G3, H_   }, { Whh[3], WhhT1[1], G3, H_   },
    };
    for (int i = 0; i < 8; ++i) {
        const int n = tr[i].N * tr[i].K;
        k_transpose_f16<<<(n + 255) / 256, 256, 0, stream>>>(tr[i].W, tr[i].WT,
                                                             tr[i].N, tr[i].K);
    }
    {
        const int n = B_ * T_ * OMNI;
        k_f32_to_f16<<<(n + 255) / 256, 256, 0, stream>>>(omni_seq, omni_h, n);
    }

    // ---- static / attention / conv branch ----
    k_static<<<B_, 64, 0, stream>>>(static_x, sm_W, sm_b, sm_ln_g, sm_ln_b,
                                    gate_W, gate_b, phys, static_embed, p_buf);
    k_qkv<<<(B_ * PL + 255) / 256, 256, 0, stream>>>(p_buf, attn_in_W, attn_in_b, qkv);
    k_attn<<<(B_ * NP * PL + 255) / 256, 256, 0, stream>>>(qkv, ao);
    k_attn_out<<<(B_ * PL + 255) / 256, 256, 0, stream>>>(ao, attn_out_W, attn_out_b, p2);
    k_conv<<<(B_ * 96 + 255) / 256, 256, 0, stream>>>(p2, conv1_W, conv1_b,
                                                      conv2_W, conv2_b,
                                                      conv3_W, conv3_b, cfeat);

    // ---- GRU scans (WMMA) ----
    dim3 sgrid(B_ / 16, 2);
    k_gru_scan<OMNI, true, false><<<sgrid, 128, 0, stream>>>(
        omni_h, WihT0[0], WihT0[1], WhhT0[0], WhhT0[1],
        bih[0], bih[1], bhh[0], bhh[1], x1, nullptr);
    k_gru_scan<128, false, true><<<sgrid, 128, 0, stream>>>(
        x1, WihT1[0], WihT1[1], WhhT1[0], WhhT1[1],
        bih[2], bih[3], bhh[2], bhh[3], nullptr, omni_embed);

    // ---- head ----
    k_head<<<B_, 128, 0, stream>>>(omni_embed, static_embed, cfeat,
                                   head_ln_g, head_ln_b, head_W1, head_b1,
                                   head_W2, head_b2, (float*)d_out);
}
